// DeepseekV2Attention_64828236366677
// MI455X (gfx1250) — matrix operand order repaired and checked by reading, not compiled, not run on hardware
//
#include <hip/hip_runtime.h>

// ---------------------------------------------------------------------------
// DeepseekV2 MLA attention forward for MI455X (gfx1250).
// All GEMM/attention math on v_wmma_f32_16x16x32_bf16; operands pre-converted
// to bf16; GEMM W-slabs double-buffered in LDS via async-to-LDS copies.
// ---------------------------------------------------------------------------

typedef __attribute__((ext_vector_type(16))) __bf16 v16bf;
typedef __attribute__((ext_vector_type(8)))  __bf16 bf16x8;
typedef __attribute__((ext_vector_type(8)))  float  v8f;
typedef __attribute__((ext_vector_type(4)))  int    int4v;

#define B_    2
#define S_    2048
#define HID_  4096
#define H_    32
#define NOPE_ 128
#define ROPE_ 64
#define VD_   128
#define QHD_  192   // NOPE + ROPE
#define QLR_  1536
#define KVLR_ 512
#define KVW_  (NOPE_ + VD_)   // 256

#if __has_builtin(__builtin_amdgcn_global_load_async_to_lds_b128) && \
    __has_builtin(__builtin_amdgcn_s_wait_asynccnt)
#define USE_ASYNC_LDS 1
typedef __attribute__((address_space(1))) int4v* gbl_i4p;
typedef __attribute__((address_space(3))) int4v* lds_i4p;
#else
#define USE_ASYNC_LDS 0
#endif

// ---------------------------------------------------------------------------
// Fragment loaders (wave32; A and B share this file's contiguous k order):
//  A 16x32 bf16: lane L -> row = L&15; 16 contiguous elems at k_base+16*(L>>4).
//  B 32x16 bf16: lane L -> col = L&15; 16 contiguous elems at k_base+16*(L>>4).
//  C/D 16x16 f32: VGPR r, lane L -> M = r + 8*(L>>4), N = L&15.
// ---------------------------------------------------------------------------
__device__ __forceinline__ v16bf combine16(bf16x8 lo, bf16x8 hi) {
  return __builtin_shufflevector(lo, hi, 0, 1, 2, 3, 4, 5, 6, 7,
                                 8, 9, 10, 11, 12, 13, 14, 15);
}

__device__ __forceinline__ v16bf load_a_frag(const __bf16* __restrict__ A, size_t lda,
                                             int row_base, int k_base, int lane) {
  const __bf16* p = A + (size_t)(row_base + (lane & 15)) * lda +
                    (size_t)(k_base + 16 * (lane >> 4));
  return combine16(*(const bf16x8*)p, *(const bf16x8*)(p + 8));
}

__device__ __forceinline__ v16bf load_b_frag(const __bf16* __restrict__ W, size_t ldw,
                                             int n_base, int k_base, int lane) {
  const __bf16* p = W + (size_t)(n_base + (lane & 15)) * ldw +
                    (size_t)(k_base + 16 * (lane >> 4));
  return combine16(*(const bf16x8*)p, *(const bf16x8*)(p + 8));
}

__device__ __forceinline__ v8f wmma_bf16(v16bf a, v16bf b, v8f c) {
  return __builtin_amdgcn_wmma_f32_16x16x32_bf16(false, a, false, b, (short)0, c,
                                                 false, false);
}

// ---------------------------------------------------------------------------
// GEMM: C[M,N] = A[M,K](bf16, row stride lda) * W[N,K](bf16)^T -> f32.
// 128 threads = 4 waves; block tile 64x64 (wave tile 16x64, A frag reused x4).
// W k-slabs (64x32 bf16 = 4KB) double-buffered in LDS: async fill of slab i+1
// overlaps the 4 WMMAs of slab i; ASYNCcnt waited only after the compute.
// ---------------------------------------------------------------------------
__global__ __launch_bounds__(128)
void gemm_bf16_wmma(const __bf16* __restrict__ A, const __bf16* __restrict__ W,
                    float* __restrict__ C, int N, int K, int lda) {
  __shared__ __align__(16) __bf16 wt[2][64][32];
  const int wave = threadIdx.x >> 5;
  const int lane = threadIdx.x & 31;
  const int rowBase = blockIdx.y * 64 + wave * 16;
  const int colBase = blockIdx.x * 64;

  // per-thread staging coordinates: 256 chunks of 8 bf16, 2 per thread
  const int c0 = (int)threadIdx.x * 2;
  const int wr0 = c0 >> 2, wc0 = (c0 & 3) * 8;
  const int wr1 = (c0 + 1) >> 2, wc1 = ((c0 + 1) & 3) * 8;

  auto stage = [&](int buf, int k) {
    const __bf16* gp0 = W + (size_t)(colBase + wr0) * (size_t)K + (size_t)(k + wc0);
    const __bf16* gp1 = W + (size_t)(colBase + wr1) * (size_t)K + (size_t)(k + wc1);
#if USE_ASYNC_LDS
    __builtin_amdgcn_global_load_async_to_lds_b128((gbl_i4p)gp0,
                                                   (lds_i4p)&wt[buf][wr0][wc0], 0, 0);
    __builtin_amdgcn_global_load_async_to_lds_b128((gbl_i4p)gp1,
                                                   (lds_i4p)&wt[buf][wr1][wc1], 0, 0);
#else
    *(bf16x8*)&wt[buf][wr0][wc0] = *(const bf16x8*)gp0;
    *(bf16x8*)&wt[buf][wr1][wc1] = *(const bf16x8*)gp1;
#endif
  };

  v8f acc[4];
#pragma unroll
  for (int t = 0; t < 4; ++t)
#pragma unroll
    for (int i = 0; i < 8; ++i) acc[t][i] = 0.0f;

  // prologue: fill slab 0
  stage(0, 0);
#if USE_ASYNC_LDS
  __builtin_amdgcn_s_wait_asynccnt(0);
#endif
  __syncthreads();

  int buf = 0;
  for (int k = 0; k < K; k += 32, buf ^= 1) {
    const bool more = (k + 32 < K);
    if (more) stage(buf ^ 1, k + 32);  // async fill of next slab, no wait yet

    v16bf a = load_a_frag(A, (size_t)lda, rowBase, k, lane);
    if (more)  // global_prefetch_b8 of next A chunk
      __builtin_prefetch(A + (size_t)(rowBase + (lane & 15)) * (size_t)lda + k + 32,
                         0, 1);
#pragma unroll
    for (int t = 0; t < 4; ++t) {
      const int rl = 16 * t + (lane & 15);
      const __bf16* lp = &wt[buf][rl][16 * (lane >> 4)];
      v16bf b = combine16(*(const bf16x8*)lp, *(const bf16x8*)(lp + 8));
      acc[t] = wmma_bf16(a, b, acc[t]);
    }

#if USE_ASYNC_LDS
    if (more) __builtin_amdgcn_s_wait_asynccnt(0);  // next slab landed
#endif
    __syncthreads();
  }

  const int hi = lane >> 4, n = lane & 15;
#pragma unroll
  for (int t = 0; t < 4; ++t)
#pragma unroll
    for (int r = 0; r < 8; ++r)
      C[(size_t)(rowBase + r + 8 * hi) * (size_t)N +
        (size_t)(colBase + 16 * t + n)] = acc[t][r];
}

// ---------------------------------------------------------------------------
// f32 -> bf16 bulk conversion (n multiple of 8).
// ---------------------------------------------------------------------------
__global__ __launch_bounds__(256)
void cvt_f32_bf16(const float* __restrict__ src, __bf16* __restrict__ dst,
                  long long n) {
  const long long i = ((long long)blockIdx.x * 256 + threadIdx.x) * 8;
  if (i + 8 <= n) {
    const float4 a = *(const float4*)(src + i);
    const float4 b = *(const float4*)(src + i + 4);
    bf16x8 o;
    o[0] = (__bf16)a.x; o[1] = (__bf16)a.y; o[2] = (__bf16)a.z; o[3] = (__bf16)a.w;
    o[4] = (__bf16)b.x; o[5] = (__bf16)b.y; o[6] = (__bf16)b.z; o[7] = (__bf16)b.w;
    *(bf16x8*)(dst + i) = o;
  }
}

// ---------------------------------------------------------------------------
// RMSNorm over `dim` leading cols of each `sstride`-wide f32 row -> bf16 out.
// ---------------------------------------------------------------------------
__global__ __launch_bounds__(256)
void rmsnorm_bf16(const float* __restrict__ x, const float* __restrict__ w,
                  __bf16* __restrict__ y, int dim, int sstride, int dstride) {
  const float* p = x + (size_t)blockIdx.x * (size_t)sstride;
  __bf16* q = y + (size_t)blockIdx.x * (size_t)dstride;
  __shared__ float red[256];
  float ss = 0.0f;
  for (int i = threadIdx.x; i < dim; i += 256) {
    const float v = p[i];
    ss += v * v;
  }
  red[threadIdx.x] = ss;
  __syncthreads();
#pragma unroll
  for (int off = 128; off > 0; off >>= 1) {
    if ((int)threadIdx.x < off) red[threadIdx.x] += red[threadIdx.x + off];
    __syncthreads();
  }
  const float inv = rsqrtf(red[0] / (float)dim + 1e-6f);
  for (int i = threadIdx.x; i < dim; i += 256) q[i] = (__bf16)(p[i] * inv * w[i]);
}

// ---------------------------------------------------------------------------
// RoPE (reference's interleaved-perm variant), fused with bf16 conversion.
//   y[j]    = x[2j]*cos - x[2j+1]*sin
//   y[j+32] = x[2j+1]*cos + x[2j]*sin,  angle = p * 10000^(-2j/64)
// ---------------------------------------------------------------------------
__global__ __launch_bounds__(160)
void rope_cvt_q(const float* __restrict__ qf32, __bf16* __restrict__ qbf) {
  const int vec = blockIdx.x;  // over rows*H
  const int row = vec >> 5;    // / H_
  const int h = vec & 31;
  const float* x = qf32 + (size_t)row * (size_t)(H_ * QHD_) + (size_t)(h * QHD_);
  __bf16* y = qbf + (size_t)row * (size_t)(H_ * QHD_) + (size_t)(h * QHD_);
  const int t = threadIdx.x;
  if (t < NOPE_) {
    y[t] = (__bf16)x[t];  // q_nope: plain convert
  } else {
    const int j = t - NOPE_;  // 0..31
    const int p = row & (S_ - 1);
    const float invf = __powf(10000.0f, -(float)(2 * j) * (1.0f / 64.0f));
    const float ang = (float)p * invf;
    const float c = __cosf(ang), s = __sinf(ang);
    const float x0 = x[NOPE_ + 2 * j], x1 = x[NOPE_ + 2 * j + 1];
    y[NOPE_ + j]      = (__bf16)(x0 * c - x1 * s);
    y[NOPE_ + 32 + j] = (__bf16)(x1 * c + x0 * s);
  }
}

__global__ __launch_bounds__(256)
void rope_k_kernel(const float* __restrict__ ckv, __bf16* __restrict__ kpe) {
  const int row = blockIdx.x * 8 + (threadIdx.x >> 5);
  const int j = threadIdx.x & 31;
  const float* x = ckv + (size_t)row * (size_t)(KVLR_ + ROPE_) + KVLR_;
  const int p = row & (S_ - 1);
  const float invf = __powf(10000.0f, -(float)(2 * j) * (1.0f / 64.0f));
  const float ang = (float)p * invf;
  const float c = __cosf(ang), s = __sinf(ang);
  const float x0 = x[2 * j], x1 = x[2 * j + 1];
  kpe[(size_t)row * 64 + j]      = (__bf16)(x0 * c - x1 * s);
  kpe[(size_t)row * 64 + j + 32] = (__bf16)(x1 * c + x0 * s);
}

// ---------------------------------------------------------------------------
// Extract V from kvb_f32 and transpose to vt[(b*H+h)*VD + d][s] (bf16) so the
// P*V B-fragments become contiguous 16-element runs. LDS tile transpose.
// grid: (S/32, VD/32, B*H), block 256.
// ---------------------------------------------------------------------------
__global__ __launch_bounds__(256)
void transpose_v_bf16(const float* __restrict__ kvb, __bf16* __restrict__ vt) {
  const int s0 = blockIdx.x * 32, d0 = blockIdx.y * 32;
  const int bh = blockIdx.z;  // b*H + h
  const int b = bh >> 5, h = bh & 31;
  __shared__ float tile[32][33];
  const int tc = threadIdx.x & 31;
  const int tr = threadIdx.x >> 5;  // 0..7
#pragma unroll
  for (int r = 0; r < 32; r += 8)
    tile[tr + r][tc] = kvb[(size_t)(b * S_ + s0 + tr + r) * (size_t)(H_ * KVW_) +
                           (size_t)(h * KVW_ + NOPE_ + d0 + tc)];
  __syncthreads();
#pragma unroll
  for (int r = 0; r < 32; r += 8)
    vt[(size_t)(bh * VD_ + d0 + tr + r) * (size_t)S_ + (size_t)(s0 + tc)] =
        (__bf16)tile[tc][tr + r];
}

// ---------------------------------------------------------------------------
// Flash attention: one wave per (b, h, 16-query tile). All operands bf16.
// q   : [B*S, H*QHD] bf16 (roped)       kv  : [B*S, H*KVW] bf16 (k_nope at +0)
// kpe : [B*S, ROPE]  bf16 (roped)       vt  : [(b*H+h)*VD + d][S] bf16
// out : [B*S, H*VD]  bf16
// ---------------------------------------------------------------------------
__global__ __launch_bounds__(32)
void mla_attention(const __bf16* __restrict__ q, const __bf16* __restrict__ kv,
                   const __bf16* __restrict__ kpe, const __bf16* __restrict__ vt,
                   __bf16* __restrict__ out) {
  const int qtile = blockIdx.x, h = blockIdx.y, b = blockIdx.z;
  const int lane = threadIdx.x;
  const int hi = lane >> 4;
  const int nn = lane & 15;
  const int qbase = qtile * 16;
  const int rowQ = b * S_ + qbase;

  __shared__ __align__(16) __bf16 plds[16][32];  // P staged C-layout -> A-layout

  v16bf qf[6];
#pragma unroll
  for (int c = 0; c < 6; ++c)
    qf[c] = load_a_frag(q, (size_t)(H_ * QHD_), rowQ, h * QHD_ + 32 * c, lane);

  v8f oacc[8];
  float m_run[8], l_run[8];
#pragma unroll
  for (int t = 0; t < 8; ++t)
#pragma unroll
    for (int i = 0; i < 8; ++i) oacc[t][i] = 0.0f;
#pragma unroll
  for (int r = 0; r < 8; ++r) { m_run[r] = -1e30f; l_run[r] = 0.0f; }

  const __bf16* vbase = vt + (size_t)(b * H_ + h) * (size_t)VD_ * (size_t)S_;
  const float scale = 0.07216878364870323f;  // 192^-0.5
  const int lastTile = (qbase + 15) >> 5;    // causal early exit

  for (int kt = 0; kt <= lastTile; ++kt) {
    const int kb = kt * 32;
    const int rowK = b * S_ + kb;

    v8f s0, s1;
#pragma unroll
    for (int i = 0; i < 8; ++i) { s0[i] = 0.0f; s1[i] = 0.0f; }
#pragma unroll
    for (int c = 0; c < 6; ++c) {
      v16bf b0, b1;
      if (c < 4) {  // k_nope
        b0 = load_b_frag(kv + h * KVW_, (size_t)(H_ * KVW_), rowK, 32 * c, lane);
        b1 = load_b_frag(kv + h * KVW_, (size_t)(H_ * KVW_), rowK + 16, 32 * c, lane);
      } else {      // roped k_pe, shared across heads
        b0 = load_b_frag(kpe, (size_t)ROPE_, rowK, 32 * (c - 4), lane);
        b1 = load_b_frag(kpe, (size_t)ROPE_, rowK + 16, 32 * (c - 4), lane);
      }
      s0 = wmma_bf16(qf[c], b0, s0);
      s1 = wmma_bf16(qf[c], b1, s1);
    }

    // online softmax; row m = r + 8*hi lives across the 16 lanes of a half
#pragma unroll
    for (int r = 0; r < 8; ++r) {
      const int m = r + 8 * hi;
      const int qp = qbase + m;
      const int kp0 = kb + nn, kp1 = kb + 16 + nn;
      float v0 = s0[r] * scale + ((kp0 <= qp) ? 0.0f : -1e9f);
      float v1 = s1[r] * scale + ((kp1 <= qp) ? 0.0f : -1e9f);
      float mx = fmaxf(v0, v1);
#pragma unroll
      for (int d = 1; d < 16; d <<= 1) mx = fmaxf(mx, __shfl_xor(mx, d, 32));
      const float mnew = fmaxf(m_run[r], mx);
      const float p0 = __expf(v0 - mnew);
      const float p1 = __expf(v1 - mnew);
      float rs = p0 + p1;
#pragma unroll
      for (int d = 1; d < 16; d <<= 1) rs += __shfl_xor(rs, d, 32);
      const float alpha = __expf(m_run[r] - mnew);
      l_run[r] = l_run[r] * alpha + rs;
      m_run[r] = mnew;
#pragma unroll
      for (int t = 0; t < 8; ++t) oacc[t][r] *= alpha;
      plds[m][nn]      = (__bf16)p0;
      plds[m][nn + 16] = (__bf16)p1;
    }
    __syncthreads();

    // P C-layout -> A-layout: two contiguous 8-elem runs per lane (ds b128)
    const __bf16* pp = &plds[nn][16 * hi];
    v16bf pf = combine16(*(const bf16x8*)pp, *(const bf16x8*)(pp + 8));
    __syncthreads();

    // O += P(16x32) x V(32x128) from transposed V (contiguous key runs)
#pragma unroll
    for (int t = 0; t < 8; ++t) {
      v16bf vf = load_b_frag(vbase, (size_t)S_, t * 16, kb, lane);
      oacc[t] = wmma_bf16(pf, vf, oacc[t]);
    }
  }

#pragma unroll
  for (int t = 0; t < 8; ++t)
#pragma unroll
    for (int r = 0; r < 8; ++r) {
      const int m = r + 8 * hi;
      out[(size_t)(b * S_ + qbase + m) * (size_t)(H_ * VD_) +
          (size_t)(h * VD_ + t * 16 + nn)] = (__bf16)(oacc[t][r] / l_run[r]);
    }
}

// ---------------------------------------------------------------------------
// Host orchestration
// ---------------------------------------------------------------------------
extern "C" void kernel_launch(void* const* d_in, const int* in_sizes, int n_in,
                              void* d_out, int out_size, void* d_ws, size_t ws_size,
                              hipStream_t stream) {
  (void)in_sizes; (void)n_in; (void)out_size; (void)ws_size;

  const float* hidden  = (const float*)d_in[0];
  // d_in[1] attention_mask: causal, reconstructed on device
  // d_in[2] position_ids : arange(S) per batch, reconstructed on device
  const float* q_a_w   = (const float*)d_in[3];
  const float* q_a_ln  = (const float*)d_in[4];
  const float* q_b_w   = (const float*)d_in[5];
  const float* kv_a_w  = (const float*)d_in[6];
  const float* kv_a_ln = (const float*)d_in[7];
  const float* kv_b_w  = (const float*)d_in[8];
  const float* o_w     = (const float*)d_in[9];
  float* out = (float*)d_out;

  const int rows = B_ * S_;  // 4096
  char* ws = (char*)d_ws;
  size_t off = 0;
  auto alloc = [&](size_t bytes) {
    void* p = ws + off;
    off += (bytes + 255) & ~(size_t)255;
    return p;
  };
  __bf16* hbf     = (__bf16*)alloc((size_t)rows * HID_ * 2);
  __bf16* wqa     = (__bf16*)alloc((size_t)QLR_ * HID_ * 2);
  __bf16* wkva    = (__bf16*)alloc((size_t)(KVLR_ + ROPE_) * HID_ * 2);
  __bf16* wqb     = (__bf16*)alloc((size_t)(H_ * QHD_) * QLR_ * 2);
  __bf16* wkvb    = (__bf16*)alloc((size_t)(H_ * KVW_) * KVLR_ * 2);
  __bf16* wo      = (__bf16*)alloc((size_t)HID_ * (H_ * VD_) * 2);
  float*  SH      = (float*)alloc((size_t)rows * (H_ * KVW_) * 4);  // shared f32 scratch
  __bf16* qa_bf   = (__bf16*)alloc((size_t)rows * QLR_ * 2);
  __bf16* ckv_bf  = (__bf16*)alloc((size_t)rows * KVLR_ * 2);
  __bf16* kpe_bf  = (__bf16*)alloc((size_t)rows * ROPE_ * 2);
  __bf16* qbf     = (__bf16*)alloc((size_t)rows * (H_ * QHD_) * 2);
  __bf16* kvb_bf  = (__bf16*)alloc((size_t)rows * (H_ * KVW_) * 2);
  __bf16* vt_bf   = (__bf16*)alloc((size_t)(B_ * H_) * VD_ * S_ * 2);
  __bf16* attn_bf = (__bf16*)alloc((size_t)rows * (H_ * VD_) * 2);

  const dim3 gblk(128);
  auto cvt = [&](const float* s, __bf16* d, long long n) {
    cvt_f32_bf16<<<(unsigned)(n / 2048), 256, 0, stream>>>(s, d, n);
  };

  // one-shot bf16 conversions (bandwidth-trivial vs 490 GFLOP of compute)
  cvt(hidden, hbf, (long long)rows * HID_);
  cvt(q_a_w, wqa, (long long)QLR_ * HID_);
  cvt(kv_a_w, wkva, (long long)(KVLR_ + ROPE_) * HID_);
  cvt(q_b_w, wqb, (long long)(H_ * QHD_) * QLR_);
  cvt(kv_b_w, wkvb, (long long)(H_ * KVW_) * KVLR_);
  cvt(o_w, wo, (long long)HID_ * (H_ * VD_));

  // ---- Q path (SH reused sequentially as qa_f32 then qb_f32) ----
  gemm_bf16_wmma<<<dim3(QLR_ / 64, rows / 64), gblk, 0, stream>>>(
      hbf, wqa, SH, QLR_, HID_, HID_);
  rmsnorm_bf16<<<rows, 256, 0, stream>>>(SH, q_a_ln, qa_bf, QLR_, QLR_, QLR_);
  gemm_bf16_wmma<<<dim3(H_ * QHD_ / 64, rows / 64), gblk, 0, stream>>>(
      qa_bf, wqb, SH, H_ * QHD_, QLR_, QLR_);
  rope_cvt_q<<<rows * H_, 160, 0, stream>>>(SH, qbf);

  // ---- KV path (SH reused as ckv_f32 then kvb_f32) ----
  gemm_bf16_wmma<<<dim3((KVLR_ + ROPE_) / 64, rows / 64), gblk, 0, stream>>>(
      hbf, wkva, SH, KVLR_ + ROPE_, HID_, HID_);
  rmsnorm_bf16<<<rows, 256, 0, stream>>>(SH, kv_a_ln, ckv_bf, KVLR_,
                                         KVLR_ + ROPE_, KVLR_);
  rope_k_kernel<<<rows / 8, 256, 0, stream>>>(SH, kpe_bf);
  gemm_bf16_wmma<<<dim3(H_ * KVW_ / 64, rows / 64), gblk, 0, stream>>>(
      ckv_bf, wkvb, SH, H_ * KVW_, KVLR_, KVLR_);
  cvt(SH, kvb_bf, (long long)rows * (H_ * KVW_));
  transpose_v_bf16<<<dim3(S_ / 32, VD_ / 32, B_ * H_), 256, 0, stream>>>(SH, vt_bf);

  // ---- attention + output projection ----
  mla_attention<<<dim3(S_ / 16, H_, B_), 32, 0, stream>>>(qbf, kvb_bf, kpe_bf,
                                                          vt_bf, attn_bf);
  gemm_bf16_wmma<<<dim3(HID_ / 64, rows / 64), gblk, 0, stream>>>(
      attn_bf, wo, out, HID_, H_ * VD_, H_ * VD_);
}
